// TopkMasker_25666724561375
// MI455X (gfx1250) — matrix operand order repaired and checked by reading, not compiled, not run on hardware
//
#include <hip/hip_runtime.h>
#include <stdint.h>

#define Bsz 2
#define Tt 1024
#define Jj 64
#define Dd 1024
#define Hh 256
#define H3 768
#define LCTX 1024

typedef __attribute__((ext_vector_type(16))) __bf16 v16bf;
typedef __attribute__((ext_vector_type(8)))  float  v8f;

union Frag { uint32_t u[8]; v16bf v; };

__device__ __forceinline__ uint16_t f2bf(float f) {
    uint32_t u = __float_as_uint(f);
    u += 0x7FFFu + ((u >> 16) & 1u);   // round-to-nearest-even
    return (uint16_t)(u >> 16);
}

__device__ __forceinline__ float gelu_exact(float v) {
    return 0.5f * v * (1.0f + erff(v * 0.70710678118654752f));
}

// ---------------------------------------------------------------------------
// K0: convert Wu/Wv (f32, row-major [h][c]) to bf16, same layout.
// ---------------------------------------------------------------------------
__global__ void cvt_w(const float* __restrict__ wu, const float* __restrict__ wv,
                      uint16_t* __restrict__ ou, uint16_t* __restrict__ ov, int n) {
    int i = blockIdx.x * 256 + threadIdx.x;
    if (i < n) {
        ou[i] = f2bf(wu[i]);
        ov[i] = f2bf(wv[i]);
    }
}

// ---------------------------------------------------------------------------
// K1/K2: Y[row][h] = dot(X[row][:], W[h][:]), D = 1024, H = 256.
// One block per row, one thread per output h. ~1 GFLOP total — negligible.
// ---------------------------------------------------------------------------
__global__ __launch_bounds__(256)
void proj(const float* __restrict__ X, const float* __restrict__ W,
          float* __restrict__ Y) {
    const int row = blockIdx.x;
    const int h   = threadIdx.x;
    const float4* x4 = (const float4*)(X + (size_t)row * Dd);
    const float4* w4 = (const float4*)(W + (size_t)h * Dd);
    float acc = 0.f;
    #pragma unroll 8
    for (int i = 0; i < Dd / 4; ++i) {
        float4 a = x4[i], b = w4[i];
        acc = fmaf(a.x, b.x, acc);
        acc = fmaf(a.y, b.y, acc);
        acc = fmaf(a.z, b.z, acc);
        acc = fmaf(a.w, b.w, acc);
    }
    Y[(size_t)row * Hh + h] = acc;
}

// ---------------------------------------------------------------------------
// K3: dominant kernel. One block = (b, 16-row t-tile, j).
//   x = [q,k,q*k] (768) -> LayerNorm -> bf16 X in LDS
//   U = X*Wu^T, V = X*Wv^T via v_wmma_f32_16x16x32_bf16 (8 waves x 2 n-tiles)
//   logit = Wo . (U * gelu(V)) + bo  via shfl reduction + ds_add_f32
// ---------------------------------------------------------------------------
__global__ __launch_bounds__(256)
void mlp_logits(const float* __restrict__ qws, const float* __restrict__ kws,
                const uint16_t* __restrict__ WuB, const uint16_t* __restrict__ WvB,
                const float* __restrict__ ln_g, const float* __restrict__ ln_b,
                const float* __restrict__ bu, const float* __restrict__ bv,
                const float* __restrict__ Wo, const float* __restrict__ bo,
                float* __restrict__ logits) {
    __shared__ float    q_lds[16 * Hh];      // 16 KB
    __shared__ float    k_lds[Hh];           // 1 KB
    __shared__ uint16_t x_lds[16 * H3];      // 24 KB bf16
    __shared__ float    red_s[16][16];
    __shared__ float    red_q[16][16];
    __shared__ float    mean_s[16], istd_s[16];
    __shared__ float    lacc[16];

    const int tid = threadIdx.x;
    const int j   = blockIdx.x & (Jj - 1);
    const int tt  = (blockIdx.x >> 6) & 63;
    const int b   = blockIdx.x >> 12;
    const int t0  = tt * 16;

    // stage q tile (16x256) and k row (256)
    const float* qg = qws + (size_t)(b * Tt + t0) * Hh;
    const float* kg = kws + (size_t)(b * Jj + j) * Hh;
    #pragma unroll
    for (int i = 0; i < 16; ++i) q_lds[i * Hh + tid] = qg[i * Hh + tid];
    k_lds[tid] = kg[tid];
    if (tid < 16) lacc[tid] = 0.0f;
    __syncthreads();

    const int row  = tid >> 4;   // 0..15
    const int part = tid & 15;
    const float* qr = &q_lds[row * Hh];

    // LayerNorm statistics over the virtual 768-vector x
    float s = 0.f, sq = 0.f;
    for (int c = part; c < H3; c += 16) {
        float x = (c < 256) ? qr[c]
                : (c < 512) ? k_lds[c - 256]
                            : qr[c - 512] * k_lds[c - 512];
        s += x; sq += x * x;
    }
    red_s[row][part] = s; red_q[row][part] = sq;
    __syncthreads();
    if (part == 0) {
        float ts = 0.f, tq = 0.f;
        #pragma unroll
        for (int i = 0; i < 16; ++i) { ts += red_s[row][i]; tq += red_q[row][i]; }
        float mu  = ts * (1.0f / H3);
        float var = tq * (1.0f / H3) - mu * mu;
        mean_s[row] = mu;
        istd_s[row] = rsqrtf(var + 1e-5f);
    }
    __syncthreads();
    {
        const float mu = mean_s[row], is = istd_s[row];
        for (int c = part; c < H3; c += 16) {
            float x = (c < 256) ? qr[c]
                    : (c < 512) ? k_lds[c - 256]
                                : qr[c - 512] * k_lds[c - 512];
            float xn = (x - mu) * is * ln_g[c] + ln_b[c];
            x_lds[row * H3 + c] = f2bf(xn);
        }
    }
    __syncthreads();

    // --- WMMA phase: 8 waves, each owns n-tiles {2w, 2w+1}, both U and V ---
    const int lane = tid & 31;
    const int wave = tid >> 5;
    const int m  = lane & 15;
    const int hl = lane >> 4;                 // lane-half
    const int nA = wave * 32 + (lane & 15);   // n index for tile 0
    const int nB = nA + 16;                   // n index for tile 1

    v8f aU0 = {}, aU1 = {}, aV0 = {}, aV1 = {};
    const uint16_t* wuA = WuB + (size_t)nA * H3;
    const uint16_t* wuB2 = WuB + (size_t)nB * H3;
    const uint16_t* wvA = WvB + (size_t)nA * H3;
    const uint16_t* wvB2 = WvB + (size_t)nB * H3;
    const uint16_t* xr = &x_lds[m * H3];

    for (int ks = 0; ks < H3; ks += 32) {
        Frag a, fu0, fu1, fv0, fv1;
        #pragma unroll
        for (int p = 0; p < 8; ++p) {
            // A (16x32 bf16): lanes0-15 K pairs {0..7,16..23}, lanes16-31 +8
            int ka = ks + ((p < 4) ? 2 * p : 2 * p + 8) + 8 * hl;
            a.u[p] = *(const uint32_t*)&xr[ka];
            // B (32x16 bf16): lanes0-15 K pairs {0..7,16..23}, lanes16-31 +8
            int kb = ks + ((p < 4) ? 2 * p : 2 * p + 8) + 8 * hl;
            fu0.u[p] = *(const uint32_t*)&wuA[kb];
            fu1.u[p] = *(const uint32_t*)&wuB2[kb];
            fv0.u[p] = *(const uint32_t*)&wvA[kb];
            fv1.u[p] = *(const uint32_t*)&wvB2[kb];
        }
        aU0 = __builtin_amdgcn_wmma_f32_16x16x32_bf16(false, a.v, false, fu0.v, (short)0, aU0, false, false);
        aU1 = __builtin_amdgcn_wmma_f32_16x16x32_bf16(false, a.v, false, fu1.v, (short)0, aU1, false, false);
        aV0 = __builtin_amdgcn_wmma_f32_16x16x32_bf16(false, a.v, false, fv0.v, (short)0, aV0, false, false);
        aV1 = __builtin_amdgcn_wmma_f32_16x16x32_bf16(false, a.v, false, fv1.v, (short)0, aV1, false, false);
    }

    // epilogue: Z = (U+bu) * gelu(V+bv); logit = sum_n Z*Wo[n]
    const float buA = bu[nA], buB = bu[nB];
    const float bvA = bv[nA], bvB = bv[nB];
    const float woA = Wo[nA], woB = Wo[nB];

    #pragma unroll
    for (int r = 0; r < 8; ++r) {   // D element r -> matrix row r + 8*hl
        float zA  = (aU0[r] + buA) * gelu_exact(aV0[r] + bvA);
        float zB  = (aU1[r] + buB) * gelu_exact(aV1[r] + bvB);
        float val = zA * woA + zB * woB;
        val += __shfl_xor(val, 1, 16);
        val += __shfl_xor(val, 2, 16);
        val += __shfl_xor(val, 4, 16);
        val += __shfl_xor(val, 8, 16);
        if ((lane & 15) == 0) atomicAdd(&lacc[r + 8 * hl], val);
    }
    __syncthreads();
    if (tid < 16)
        logits[(size_t)(b * Tt + t0 + tid) * Jj + j] = lacc[tid] + bo[0];
}

// ---------------------------------------------------------------------------
// K4: per (b,t) row: softmax over 64 logits, sink(5)+top-11 causal selection,
// then 1024-wide bias with triangular -inf mask.
// ---------------------------------------------------------------------------
__global__ __launch_bounds__(256)
void bias_kernel(const float* __restrict__ logits, float* __restrict__ bias) {
    __shared__ float lg[Jj];
    __shared__ float pcb[Jj];
    __shared__ int   sel[Jj];
    const int tid = threadIdx.x;
    const int t   = blockIdx.x & (Tt - 1);
    const int b   = blockIdx.x >> 10;

    const float* lrow = logits + (size_t)(b * Tt + t) * Jj;
    if (tid < Jj) { lg[tid] = lrow[tid]; sel[tid] = (tid < 5) ? 1 : 0; }
    __syncthreads();

    if (tid == 0) {
        float mx = -3.0e38f;
        for (int k = 0; k < Jj; ++k) mx = fmaxf(mx, lg[k]);
        float sum = 0.f;
        for (int k = 0; k < Jj; ++k) sum += expf(lg[k] - mx);
        const float inv = 1.0f / sum;
        for (int k = 0; k < Jj; ++k)
            pcb[k] = -8.0f * (1.0f - expf(lg[k] - mx) * inv);   // -ALPHA*(1-w_soft)

        // top-11 over tail j in [5,63] with causal mask (j <= t+5); stable ties
        const int limit = (t + 5 < 63) ? (t + 5) : 63;
        for (int it = 0; it < 11; ++it) {
            float best = 0.f; int bi = -1;
            for (int k = 5; k < Jj; ++k) {
                if (!sel[k]) {
                    float v = (k <= limit) ? lg[k] : -INFINITY;
                    if (bi < 0 || v > best) { best = v; bi = k; }
                }
            }
            if (bi >= 0) sel[bi] = 1;
        }
    }
    __syncthreads();

    float* orow = bias + (size_t)(b * Tt + t) * LCTX;
    for (int col = tid; col < LCTX; col += 256) {
        int c = col >> 4;                       // chunk_len = 16
        float v = (col > t) ? -INFINITY : (sel[c] ? 0.0f : pcb[c]);
        orow[col] = v;
    }
}

// ---------------------------------------------------------------------------
extern "C" void kernel_launch(void* const* d_in, const int* in_sizes, int n_in,
                              void* d_out, int out_size, void* d_ws, size_t ws_size,
                              hipStream_t stream) {
    const float* Q    = (const float*)d_in[0];
    const float* K    = (const float*)d_in[1];
    const float* Wq   = (const float*)d_in[2];
    const float* Wk   = (const float*)d_in[3];
    const float* ln_g = (const float*)d_in[4];
    const float* ln_b = (const float*)d_in[5];
    const float* Wu   = (const float*)d_in[6];
    const float* bu   = (const float*)d_in[7];
    const float* Wv   = (const float*)d_in[8];
    const float* bv   = (const float*)d_in[9];
    const float* Wo   = (const float*)d_in[10];
    const float* bo   = (const float*)d_in[11];
    (void)in_sizes; (void)n_in; (void)out_size; (void)ws_size;

    // workspace layout
    char* ws = (char*)d_ws;
    uint16_t* WuB = (uint16_t*)ws;                              // 393216 B
    uint16_t* WvB = (uint16_t*)(ws + 393216);                   // 393216 B
    float*    qws = (float*)(ws + 786432);                      // 2 MB
    float*    kws = (float*)(ws + 786432 + 2097152);            // 128 KB

    float* logits = (float*)d_out;                              // (B,T,J)
    float* biasO  = logits + (size_t)Bsz * Tt * Jj;             // (B,T,LCTX)

    hipLaunchKernelGGL(cvt_w, dim3((Hh * H3 + 255) / 256), dim3(256), 0, stream,
                       Wu, Wv, WuB, WvB, Hh * H3);
    hipLaunchKernelGGL(proj, dim3(Bsz * Tt), dim3(256), 0, stream, Q, Wq, qws);
    hipLaunchKernelGGL(proj, dim3(Bsz * Jj), dim3(256), 0, stream, K, Wk, kws);
    hipLaunchKernelGGL(mlp_logits, dim3(Bsz * (Tt / 16) * Jj), dim3(256), 0, stream,
                       qws, kws, WuB, WvB, ln_g, ln_b, bu, bv, Wo, bo, logits);
    hipLaunchKernelGGL(bias_kernel, dim3(Bsz * Tt), dim3(256), 0, stream,
                       logits, biasO);
}
